// StreamingTransformerEncoder_56186762166328
// MI455X (gfx1250) — matrix operand order repaired and checked by reading, not compiled, not run on hardware
//
#include <hip/hip_runtime.h>
#include <hip/hip_bf16.h>
#include <math.h>

#define DIM      1024
#define HEADS    16
#define HEAD_DIM 64
#define LAYERS   6
#define HID      4096
#define TSEQ     2048
#define BATCH    2
#define PASTCTX  1000
#define TKEY     (TSEQ + 1)        // 2049 (past row + T)
#define MROWS    (BATCH * TSEQ)    // 4096
#define KVROWS   (BATCH * TKEY)    // 4098

typedef unsigned short U16;
typedef unsigned int   U32;

typedef __bf16 bf16_t;
typedef bf16_t v16bf __attribute__((ext_vector_type(16)));
typedef float  v8f   __attribute__((ext_vector_type(8)));

union Frag {
  v16bf v;
  U16   u[16];
  uint4 q[2];
};

__device__ __forceinline__ U16 f2bf(float f) {
  union { float f; U32 u; } c; c.f = f;
  U32 u = c.u;
  u = (u + 0x7FFFu + ((u >> 16) & 1u)) >> 16;   // round-to-nearest-even
  return (U16)u;
}

__device__ __forceinline__ v8f wmma_bf16(const Frag& a, const Frag& b, v8f c) {
  return __builtin_amdgcn_wmma_f32_16x16x32_bf16(
      /*neg_a=*/false, a.v, /*neg_b=*/false, b.v,
      /*c_mod=*/(short)0, c, /*reuse_a=*/false, /*reuse_b=*/false);
}

// ---------------------------------------------------------------------------
// Generic GEMM:  Y[M,N] = A_bf16[M,K] @ W_f32[N,K]^T + bias[N]  (+optional GELU)
// Block: 256 thr = 8 waves; tile 128x128; each wave owns a 32x64 register tile
// (acc[2][4] -> 8 WMMAs / K-step).  Global->register double buffering hides
// HBM latency behind the WMMAs; weight stream prefetched 4 tiles ahead.
// ---------------------------------------------------------------------------
__global__ __launch_bounds__(256) void gemm_bf16_wmma(
    const U16* __restrict__ A, const float* __restrict__ W,
    const float* __restrict__ bias, float* __restrict__ Yf,
    U16* __restrict__ Yb, int M, int N, int K, int act)
{
  __shared__ U16 As[128][32];   // 8 KB
  __shared__ U16 Bs[128][32];   // 8 KB
  const int tid  = threadIdx.x;
  const int wave = tid >> 5, lane = tid & 31;
  const int wr = wave >> 1, wc = wave & 1;        // 4x2 wave grid
  const int m0 = blockIdx.y * 128, n0 = blockIdx.x * 128;
  const int fm = lane & 15, fh = lane >> 4;

  uint4  aReg[2];
  float4 wReg[4];

  auto loadA = [&](int k0) {
    #pragma unroll
    for (int i = 0; i < 2; ++i) {
      int idx = tid + i * 256;
      int r = idx >> 2;
      int c = (idx & 3) * 8;
      int gr = m0 + r; if (gr > M - 1) gr = M - 1;
      aReg[i] = *(const uint4*)(A + (size_t)gr * K + k0 + c);
    }
  };
  auto loadW = [&](int k0) {
    #pragma unroll
    for (int i = 0; i < 4; ++i) {
      int idx = tid + i * 256;
      int r = idx >> 3;
      int c = (idx & 7) * 4;
      wReg[i] = *(const float4*)(W + (size_t)(n0 + r) * K + k0 + c);
    }
  };

  loadA(0);
  loadW(0);

  v8f acc[2][4] = {};

  for (int k0 = 0; k0 < K; k0 += 32) {
    // commit staged registers into LDS (A already bf16; W converted f32->bf16)
    #pragma unroll
    for (int i = 0; i < 2; ++i) {
      int idx = tid + i * 256;
      int r = idx >> 2;
      int c = (idx & 3) * 8;
      *(uint4*)&As[r][c] = aReg[i];
    }
    #pragma unroll
    for (int i = 0; i < 4; ++i) {
      int idx = tid + i * 256;
      int r = idx >> 3;
      int c = (idx & 7) * 4;
      Bs[r][c + 0] = f2bf(wReg[i].x);
      Bs[r][c + 1] = f2bf(wReg[i].y);
      Bs[r][c + 2] = f2bf(wReg[i].z);
      Bs[r][c + 3] = f2bf(wReg[i].w);
    }
    __syncthreads();

    // start fetching next K-tile while this one is consumed by WMMA
    if (k0 + 32 < K) {
      loadA(k0 + 32);
      loadW(k0 + 32);
    }
    // stream weights ahead into L2 (global_prefetch_b8)
    if (k0 + 128 < K) {
      __builtin_prefetch(W + (size_t)(n0 + (tid >> 1)) * K + k0 + 128 +
                             (tid & 1) * 16, 0, 1);
    }

    Frag af[2], bf[4];
    #pragma unroll
    for (int i = 0; i < 2; ++i) {
      int row = wr * 32 + i * 16 + fm;
      af[i].q[0] = *(const uint4*)&As[row][fh * 8];        // K = h*8..h*8+7
      af[i].q[1] = *(const uint4*)&As[row][16 + fh * 8];   // K = 16+h*8..
    }
    #pragma unroll
    for (int j = 0; j < 4; ++j) {
      int col = wc * 64 + j * 16 + fm;
      bf[j].q[0] = *(const uint4*)&Bs[col][fh * 8];        // K = h*8..h*8+7
      bf[j].q[1] = *(const uint4*)&Bs[col][16 + fh * 8];   // K = 16+h*8..
    }
    #pragma unroll
    for (int i = 0; i < 2; ++i)
      #pragma unroll
      for (int j = 0; j < 4; ++j)
        acc[i][j] = wmma_bf16(af[i], bf[j], acc[i][j]);
    __syncthreads();
  }

  #pragma unroll
  for (int i = 0; i < 2; ++i)
    #pragma unroll
    for (int j = 0; j < 4; ++j)
      #pragma unroll
      for (int v = 0; v < 8; ++v) {
        int row = m0 + wr * 32 + i * 16 + v + 8 * fh;
        int col = n0 + wc * 64 + j * 16 + fm;
        if (row < M) {
          float y = acc[i][j][v] + bias[col];
          if (act == 1) y = 0.5f * y * (1.0f + erff(y * 0.70710678118654752f));
          size_t o = (size_t)row * N + col;
          if (Yf) Yf[o] = y;
          if (Yb) Yb[o] = f2bf(y);
        }
      }
}

// ---------------------------------------------------------------------------
// Flash-style sliding-window attention, one wave per (qtile, head, batch).
// scores = QK^T/8 with mask (0 <= qpos-k <= 1000), online softmax, P@V.
// ---------------------------------------------------------------------------
__global__ __launch_bounds__(32) void attn_wmma(
    const U16* __restrict__ Q, const U16* __restrict__ KV, U16* __restrict__ O)
{
  const int qt = blockIdx.x;     // 0..127
  const int h  = blockIdx.y;     // 0..15
  const int b  = blockIdx.z;     // 0..1
  const int lane = threadIdx.x;
  const int n = lane & 15, fh = lane >> 4;
  const int q0 = qt * 16;

  const U16* Qb = Q  + (size_t)b * TSEQ * DIM + h * HEAD_DIM;       // stride DIM
  const U16* Kb = KV + (size_t)b * TKEY * (2 * DIM) + h * HEAD_DIM; // stride 2*DIM
  const U16* Vb = Kb + DIM;

  __shared__ U16 plds[16][32];
  __shared__ U16 vlds[32][64];

  // Q A-fragments (rows q0..q0+15, K-dim = 64 split into two 16x32 chunks)
  Frag aq0, aq1;
  {
    const U16* qr = Qb + (size_t)(q0 + n) * DIM;
    aq0.q[0] = *(const uint4*)(qr + fh * 8);
    aq0.q[1] = *(const uint4*)(qr + 16 + fh * 8);
    aq1.q[0] = *(const uint4*)(qr + 32 + fh * 8);
    aq1.q[1] = *(const uint4*)(qr + 48 + fh * 8);
  }

  float rm[8], rl[8];
  v8f acc[4] = {};
  #pragma unroll
  for (int v = 0; v < 8; ++v) { rm[v] = -1e30f; rl[v] = 0.f; }

  int kmin = q0 + 1 - PASTCTX; if (kmin < 0) kmin = 0;
  int kmax = q0 + 16; if (kmax > TKEY - 1) kmax = TKEY - 1;

  for (int kc = kmin & ~31; kc <= kmax; kc += 32) {
    // stage V[kc..kc+31][0..63] into LDS (one row per lane)
    {
      int kr = kc + lane; if (kr > TKEY - 1) kr = TKEY - 1;
      const U16* vr = Vb + (size_t)kr * (2 * DIM);
      #pragma unroll
      for (int j = 0; j < 8; ++j)
        *(uint4*)&vlds[lane][j * 8] = *(const uint4*)(vr + j * 8);
    }
    __syncthreads();

    // K B-fragments (element (d, key) = K[kc+key][d]), two 16-key tiles
    int kr0 = kc + n;      if (kr0 > TKEY - 1) kr0 = TKEY - 1;
    int kr1 = kc + 16 + n; if (kr1 > TKEY - 1) kr1 = TKEY - 1;
    const U16* kp0 = Kb + (size_t)kr0 * (2 * DIM);
    const U16* kp1 = Kb + (size_t)kr1 * (2 * DIM);
    Frag b0lo, b0hi, b1lo, b1hi;
    b0lo.q[0] = *(const uint4*)(kp0 + fh * 8);
    b0lo.q[1] = *(const uint4*)(kp0 + 16 + fh * 8);
    b0hi.q[0] = *(const uint4*)(kp0 + 32 + fh * 8);
    b0hi.q[1] = *(const uint4*)(kp0 + 48 + fh * 8);
    b1lo.q[0] = *(const uint4*)(kp1 + fh * 8);
    b1lo.q[1] = *(const uint4*)(kp1 + 16 + fh * 8);
    b1hi.q[0] = *(const uint4*)(kp1 + 32 + fh * 8);
    b1hi.q[1] = *(const uint4*)(kp1 + 48 + fh * 8);

    v8f s0 = {}, s1 = {};
    s0 = wmma_bf16(aq0, b0lo, s0);
    s0 = wmma_bf16(aq1, b0hi, s0);
    s1 = wmma_bf16(aq0, b1lo, s1);
    s1 = wmma_bf16(aq1, b1hi, s1);

    // mask + row max (rows live across a 16-lane group)
    float mloc[8];
    #pragma unroll
    for (int v = 0; v < 8; ++v) {
      int row = v + 8 * fh;
      int qpos = q0 + row + 1;
      float x0 = s0[v] * 0.125f;           // 1/sqrt(64)
      float x1 = s1[v] * 0.125f;
      int k0i = kc + n, k1i = kc + 16 + n;
      bool ok0 = (k0i <= qpos) && (qpos - k0i <= PASTCTX) && (k0i < TKEY);
      bool ok1 = (k1i <= qpos) && (qpos - k1i <= PASTCTX) && (k1i < TKEY);
      x0 = ok0 ? x0 : -1e30f;
      x1 = ok1 ? x1 : -1e30f;
      s0[v] = x0; s1[v] = x1;
      mloc[v] = fmaxf(x0, x1);
    }
    #pragma unroll
    for (int off = 1; off < 16; off <<= 1)
      #pragma unroll
      for (int v = 0; v < 8; ++v)
        mloc[v] = fmaxf(mloc[v], __shfl_xor(mloc[v], off, 16));

    // online softmax update + emit P tile to LDS
    float psum[8];
    #pragma unroll
    for (int v = 0; v < 8; ++v) {
      float mn = fmaxf(rm[v], mloc[v]);
      float sc = __expf(rm[v] - mn);
      rm[v] = mn;
      rl[v] *= sc;
      #pragma unroll
      for (int nt = 0; nt < 4; ++nt) acc[nt][v] *= sc;
      float p0 = (s0[v] > -1e29f) ? __expf(s0[v] - mn) : 0.f;
      float p1 = (s1[v] > -1e29f) ? __expf(s1[v] - mn) : 0.f;
      psum[v] = p0 + p1;
      int row = v + 8 * fh;
      plds[row][n]      = f2bf(p0);
      plds[row][16 + n] = f2bf(p1);
    }
    #pragma unroll
    for (int off = 1; off < 16; off <<= 1)
      #pragma unroll
      for (int v = 0; v < 8; ++v)
        psum[v] += __shfl_xor(psum[v], off, 16);
    #pragma unroll
    for (int v = 0; v < 8; ++v) rl[v] += psum[v];

    __syncthreads();

    // P A-fragment from LDS, V B-fragments (element (k,dim) = V[kc+k][dim])
    Frag pa;
    pa.q[0] = *(const uint4*)&plds[n][fh * 16];
    pa.q[1] = *(const uint4*)&plds[n][fh * 16 + 8];
    #pragma unroll
    for (int nt = 0; nt < 4; ++nt) {
      Frag bv;
      #pragma unroll
      for (int j = 0; j < 16; ++j)
        bv.u[j] = vlds[fh * 16 + j][nt * 16 + n];
      acc[nt] = wmma_bf16(pa, bv, acc[nt]);
    }
    __syncthreads();
  }

  #pragma unroll
  for (int v = 0; v < 8; ++v) rl[v] = (rl[v] > 0.f) ? (1.0f / rl[v]) : 0.f;
  #pragma unroll
  for (int nt = 0; nt < 4; ++nt)
    #pragma unroll
    for (int v = 0; v < 8; ++v) {
      int row = q0 + v + 8 * fh;
      int col = h * HEAD_DIM + nt * 16 + n;
      O[(size_t)(b * TSEQ + row) * DIM + col] = f2bf(acc[nt][v] * rl[v]);
    }
}

// ---------------------------------------------------------------------------
// LayerNorm helpers / kernels
// ---------------------------------------------------------------------------
__device__ __forceinline__ void block_reduce2(float& s, float& s2) {
  __shared__ float red[16];
  #pragma unroll
  for (int off = 16; off > 0; off >>= 1) {
    s  += __shfl_xor(s, off, 32);
    s2 += __shfl_xor(s2, off, 32);
  }
  if ((threadIdx.x & 31) == 0) {
    red[threadIdx.x >> 5]       = s;
    red[8 + (threadIdx.x >> 5)] = s2;
  }
  __syncthreads();
  s = 0.f; s2 = 0.f;
  #pragma unroll
  for (int i = 0; i < 8; ++i) { s += red[i]; s2 += red[8 + i]; }
  __syncthreads();
}

__global__ __launch_bounds__(256) void embed_ln(
    const float* __restrict__ xin, const float* __restrict__ g,
    const float* __restrict__ bt, float* __restrict__ xf,
    U16* __restrict__ xb, U16* __restrict__ keysb)
{
  const int row = blockIdx.x;            // 0..4095
  const int t = row & (TSEQ - 1);
  const int bat = row >> 11;
  const float* xr = xin + (size_t)row * DIM;
  float vals[4], s = 0.f, s2 = 0.f;
  #pragma unroll
  for (int k = 0; k < 4; ++k) {
    int c = threadIdx.x + k * 256;
    float v = xr[c];
    vals[k] = v; s += v; s2 += v * v;
  }
  block_reduce2(s, s2);
  const float mean = s * (1.0f / DIM);
  const float var  = s2 * (1.0f / DIM) - mean * mean;
  const float rstd = rsqrtf(var + 1e-5f);
  const float lm = logf(10000.0f) / 511.0f;   // half-1 = 511
  #pragma unroll
  for (int k = 0; k < 4; ++k) {
    int c = threadIdx.x + k * 256;
    float y = (vals[k] - mean) * rstd * g[c] + bt[c];
    int a = (c < 512) ? c : (c - 512);
    float phase = (float)t * __expf(-(float)a * lm);
    y += (c < 512) ? __cosf(phase) : __sinf(phase);
    xf[(size_t)row * DIM + c] = y;
    U16 hv = f2bf(y);
    xb[(size_t)row * DIM + c] = hv;
    keysb[((size_t)bat * TKEY + t + 1) * DIM + c] = hv;
  }
}

__global__ __launch_bounds__(256) void resid_ln(
    const float* __restrict__ x, const float* __restrict__ d,
    const float* __restrict__ g, const float* __restrict__ bt,
    float* __restrict__ xf, U16* __restrict__ xb, U16* __restrict__ keysb)
{
  const int row = blockIdx.x;
  const int t = row & (TSEQ - 1);
  const int bat = row >> 11;
  float vals[4], s = 0.f, s2 = 0.f;
  #pragma unroll
  for (int k = 0; k < 4; ++k) {
    int c = threadIdx.x + k * 256;
    float v = x[(size_t)row * DIM + c] + d[(size_t)row * DIM + c];
    vals[k] = v; s += v; s2 += v * v;
  }
  block_reduce2(s, s2);
  const float mean = s * (1.0f / DIM);
  const float var  = s2 * (1.0f / DIM) - mean * mean;
  const float rstd = rsqrtf(var + 1e-5f);
  #pragma unroll
  for (int k = 0; k < 4; ++k) {
    int c = threadIdx.x + k * 256;
    float y = (vals[k] - mean) * rstd * g[c] + bt[c];
    xf[(size_t)row * DIM + c] = y;
    if (xb)    xb[(size_t)row * DIM + c] = f2bf(y);
    if (keysb) keysb[((size_t)bat * TKEY + t + 1) * DIM + c] = f2bf(y);
  }
}

__global__ void zero_keys(U16* keysb) {
  U16* p = keysb + (size_t)blockIdx.x * TKEY * DIM;   // past row per batch
  for (int i = threadIdx.x; i < DIM; i += blockDim.x) p[i] = 0;
}

// ---------------------------------------------------------------------------
extern "C" void kernel_launch(void* const* d_in, const int* in_sizes, int n_in,
                              void* d_out, int out_size, void* d_ws, size_t ws_size,
                              hipStream_t stream) {
  const float* x_in  = (const float*)d_in[0];
  const float* nin_g = (const float*)d_in[1];
  const float* nin_b = (const float*)d_in[2];
  const float* Wqkv  = (const float*)d_in[3];
  const float* bqkv  = (const float*)d_in[4];
  const float* Wo    = (const float*)d_in[5];
  const float* bo    = (const float*)d_in[6];
  const float* ln1g  = (const float*)d_in[7];
  const float* ln1b  = (const float*)d_in[8];
  const float* ln2g  = (const float*)d_in[9];
  const float* ln2b  = (const float*)d_in[10];
  const float* W1    = (const float*)d_in[11];
  const float* b1    = (const float*)d_in[12];
  const float* W2    = (const float*)d_in[13];
  const float* b2    = (const float*)d_in[14];
  (void)in_sizes; (void)n_in; (void)out_size; (void)ws_size;

  char* ws = (char*)d_ws;
  size_t off = 0;
  auto carve = [&](size_t bytes) -> char* {
    char* p = ws + off;
    off += (bytes + 255) & ~(size_t)255;
    return p;
  };
  float* xf    = (float*)carve((size_t)MROWS * DIM * 4);        // activations f32
  U16*   xb    = (U16*)  carve((size_t)MROWS * DIM * 2);        // activations bf16
  U16*   keysb = (U16*)  carve((size_t)KVROWS * DIM * 2);       // [past|x] bf16
  U16*   qb    = (U16*)  carve((size_t)MROWS * DIM * 2);        // Q bf16
  U16*   kvb   = (U16*)  carve((size_t)KVROWS * 2 * DIM * 2);   // K|V bf16
  U16*   ob    = (U16*)  carve((size_t)MROWS * DIM * 2);        // attn out bf16
  U16*   hb    = (U16*)  carve((size_t)MROWS * HID * 2);        // FF hidden bf16
  float* delta = (float*)carve((size_t)MROWS * DIM * 4);        // residual delta

  embed_ln<<<MROWS, 256, 0, stream>>>(x_in, nin_g, nin_b, xf, xb, keysb);
  zero_keys<<<BATCH, 256, 0, stream>>>(keysb);

  for (int l = 0; l < LAYERS; ++l) {
    const float* Wqkv_l = Wqkv + (size_t)l * 3 * DIM * DIM;
    const float* bqkv_l = bqkv + (size_t)l * 3 * DIM;
    const float* Wo_l   = Wo + (size_t)l * DIM * DIM;
    const float* bo_l   = bo + (size_t)l * DIM;
    const float* W1_l   = W1 + (size_t)l * HID * DIM;
    const float* b1_l   = b1 + (size_t)l * HID;
    const float* W2_l   = W2 + (size_t)l * DIM * HID;
    const float* b2_l   = b2 + (size_t)l * DIM;

    // Q = x @ Wq^T + bq
    gemm_bf16_wmma<<<dim3(DIM / 128, MROWS / 128), 256, 0, stream>>>(
        xb, Wqkv_l, bqkv_l, nullptr, qb, MROWS, DIM, DIM, 0);
    // [K|V] = keys @ [Wk;Wv]^T + [bk;bv]
    gemm_bf16_wmma<<<dim3(2 * DIM / 128, (KVROWS + 127) / 128), 256, 0, stream>>>(
        keysb, Wqkv_l + (size_t)DIM * DIM, bqkv_l + DIM, nullptr, kvb,
        KVROWS, 2 * DIM, DIM, 0);
    attn_wmma<<<dim3(TSEQ / 16, HEADS, BATCH), 32, 0, stream>>>(qb, kvb, ob);
    // o @ Wo^T + bo -> delta (f32)
    gemm_bf16_wmma<<<dim3(DIM / 128, MROWS / 128), 256, 0, stream>>>(
        ob, Wo_l, bo_l, delta, nullptr, MROWS, DIM, DIM, 0);
    // x = ln1(x + o)
    resid_ln<<<MROWS, 256, 0, stream>>>(xf, delta, ln1g + l * DIM, ln1b + l * DIM,
                                        xf, xb, nullptr);
    // h = gelu(x @ W1^T + b1) (bf16)
    gemm_bf16_wmma<<<dim3(HID / 128, MROWS / 128), 256, 0, stream>>>(
        xb, W1_l, b1_l, nullptr, hb, MROWS, HID, DIM, 1);
    // ff = h @ W2^T + b2 -> delta (f32)
    gemm_bf16_wmma<<<dim3(DIM / 128, MROWS / 128), 256, 0, stream>>>(
        hb, W2_l, b2_l, delta, nullptr, MROWS, DIM, HID, 0);
    // x = ln2(x + ff); final layer writes d_out directly
    const bool last = (l == LAYERS - 1);
    resid_ln<<<MROWS, 256, 0, stream>>>(xf, delta, ln2g + l * DIM, ln2b + l * DIM,
                                        last ? (float*)d_out : xf,
                                        last ? nullptr : xb,
                                        last ? nullptr : keysb);
  }
}
